// _FasterRCNNHead_74629351735608
// MI455X (gfx1250) — matrix operand order repaired and checked by reading, not compiled, not run on hardware
//
#include <hip/hip_runtime.h>
#include <hip/hip_bf16.h>

typedef __attribute__((ext_vector_type(16))) _Float16 v16h;
typedef __attribute__((ext_vector_type(8)))  _Float16 v8h;
typedef __attribute__((ext_vector_type(8)))  float    v8f;

#define NEG_INF (-__builtin_inff())
#define IMG_F 800.0f
#define BBOX_CLIP_F 4.135166556742356f
#define TOT_PIX 53125
#define TOT_PROP 159375
#define POST_NMS 2000
#define DETS 100

// ---------------------------------------------------------------------------
// rpn_conv_w f32 [oc][c][3][3]  ->  f16 [tap][oc][c]  (B-fragment friendly)
// ---------------------------------------------------------------------------
__global__ void cvt_rpnw(const float* __restrict__ w, _Float16* __restrict__ wt) {
  int i = blockIdx.x * blockDim.x + threadIdx.x;
  if (i >= 9 * 128 * 128) return;
  int tap = i / (128 * 128);
  int oc = (i / 128) % 128;
  int c = i % 128;
  wt[i] = (_Float16)w[((size_t)oc * 128 + c) * 9 + tap];
}

// ---------------------------------------------------------------------------
// NCHW f32 -> NHWC f16 feature transpose (per batch+level). Coalesced reads.
// ---------------------------------------------------------------------------
__global__ void cvt_feat_nhwc(const float* __restrict__ in,  // [128][HW]
                              _Float16* __restrict__ out,    // [HW][128]
                              int HW) {
  int i = blockIdx.x * blockDim.x + threadIdx.x;
  if (i >= 128 * HW) return;
  int c = i / HW, p = i % HW;
  out[(size_t)p * 128 + c] = (_Float16)in[i];
}

__global__ void zero_row_f16(_Float16* __restrict__ z) {
  int i = threadIdx.x;
  if (i < 128) z[i] = (_Float16)0.f;
}

// ---------------------------------------------------------------------------
// RPN 3x3 conv (C=128 -> 128), implicit GEMM via WMMA f16, fused bias+ReLU.
// NHWC f16 input: A fragment = one aligned v16h load; padding via pointer
// select to a zero row (cndmask on address, no EXEC branching, no masking ALU).
// One wave per 16-pixel x 16-outchannel tile. K = 9 taps x 128 ch = 36 wmma.
// ---------------------------------------------------------------------------
__global__ void rpn_conv3x3_wmma(const _Float16* __restrict__ tin,  // [HW][128] f16
                                 const _Float16* __restrict__ wt,   // [9][128][128] f16
                                 const float* __restrict__ bias,    // [128]
                                 const _Float16* __restrict__ zrow, // [128] zeros
                                 _Float16* __restrict__ out,        // [HW][128] f16
                                 int H, int W) {
  const int HW = H * W;
  const int lane = threadIdx.x & 31;
  const int wave = (blockIdx.x * blockDim.x + threadIdx.x) >> 5;
  const int ptiles = (HW + 15) >> 4;
  const int pt = wave % ptiles;
  const int ot = wave / ptiles;
  if (ot >= 8) return;

  const int half = lane >> 4;
  const int lm = lane & 15;
  const int p = pt * 16 + lm;
  const bool pvalid = p < HW;
  const int pp = pvalid ? p : HW - 1;
  const int y = pp / W, x = pp % W;
  const int oc = ot * 16 + lm;  // B column (n = lane&15)
  const _Float16* wbase = wt + ((size_t)oc << 7) + half * 16;  // + tap*16384 + c0

  v8f acc = {};
  for (int tap = 0; tap < 9; ++tap) {
    const int dy = tap / 3 - 1, dx = tap % 3 - 1;
    const int yy = y + dy, xx = x + dx;
    const bool inb = pvalid && yy >= 0 && yy < H && xx >= 0 && xx < W;
    const _Float16* src = inb ? (tin + ((size_t)yy * W + xx) * 128) : zrow;
    const _Float16* wtap = wbase + (size_t)tap * 16384;
    #pragma unroll
    for (int c0 = 0; c0 < 128; c0 += 32) {
      // A layout: k = half*16+j (matches B)
      v16h a = *(const v16h*)(src + c0 + half * 16);
      // B layout: lanes0-15 hold K=0..15, lanes16-31 K=16..31 -> 32B v16h load
      v16h b = *(const v16h*)(wtap + c0);
      acc = __builtin_amdgcn_wmma_f32_16x16x32_f16(false, a, false, b,
                                                   (short)0, acc, false, false);
    }
  }
  // C/D layout: lane -> N = lane&15, M = r + 8*(lane>>4)
  const int n = ot * 16 + lm;
  const int mb = pt * 16 + (half << 3);
  const float bv = bias[n];
  #pragma unroll
  for (int r = 0; r < 8; ++r) {
    int pm = mb + r;
    if (pm < HW) {
      float v = acc[r] + bv;
      out[(size_t)pm * 128 + n] = (_Float16)(v > 0.f ? v : 0.f);
    }
  }
}

// ---------------------------------------------------------------------------
// Generic f16 WMMA GEMM: Out[M,N] = act(A[M,K] @ W[N,K]^T + bias), f16 out.
// M mult of 16 (2000 = 125*16), N mult of 16, K mult of 64 (6272, 1024).
// ---------------------------------------------------------------------------
__global__ void gemm_wmma_f16(const _Float16* __restrict__ A,  // [M,K]
                              const _Float16* __restrict__ W,  // [N,K]
                              const float* __restrict__ bias,  // [N]
                              _Float16* __restrict__ Out,      // [M,N]
                              int M, int N, int K, int relu) {
  const int lane = threadIdx.x & 31;
  const int wave = (blockIdx.x * blockDim.x + threadIdx.x) >> 5;
  const int tilesN = N >> 4;
  const int tilesM = M >> 4;
  const int tm = wave / tilesN, tn = wave % tilesN;
  if (tm >= tilesM) return;
  const int half = lane >> 4;
  const int lm = lane & 15;

  const _Float16* arow = A + (size_t)(tm * 16 + lm) * K + half * 16;
  const _Float16* brow = W + (size_t)(tn * 16 + lm) * K + half * 16;
  v8f acc = {};
  #pragma unroll 2
  for (int k0 = 0; k0 < K; k0 += 32) {
    __builtin_prefetch((const void*)(arow + k0 + 512), 0, 1);
    v16h a = *(const v16h*)(arow + k0);       // k = half*16 + j
    v16h b = *(const v16h*)(brow + k0);       // k = half*16 + j (32B aligned)
    acc = __builtin_amdgcn_wmma_f32_16x16x32_f16(false, a, false, b,
                                                 (short)0, acc, false, false);
  }
  const int n = tn * 16 + lm;
  const int mb = tm * 16 + (half << 3);
  const float bv = bias[n];
  #pragma unroll
  for (int r = 0; r < 8; ++r) {
    float v = acc[r] + bv;
    if (relu) v = v > 0.f ? v : 0.f;
    Out[(size_t)(mb + r) * N + n] = (_Float16)v;
  }
}

// ---------------------------------------------------------------------------
// f32 -> f16 convert (fc weights)
// ---------------------------------------------------------------------------
__global__ void cvt_f32_f16(const float* __restrict__ in, _Float16* __restrict__ out, int n) {
  int i = blockIdx.x * blockDim.x + threadIdx.x;
  if (i < n) out[i] = (_Float16)in[i];
}

// ---------------------------------------------------------------------------
// RPN heads (1x1 convs) + anchor decode + clip + sigmoid, per proposal.
// t is NHWC: each thread's 128-ch dot reads one contiguous 256B row.
// ---------------------------------------------------------------------------
__global__ void proposal_kernel(const _Float16* __restrict__ t,  // batch base: [53125][128]
                                const float* __restrict__ clsw, const float* __restrict__ clsb,
                                const float* __restrict__ bbw,  const float* __restrict__ bbb,
                                float* __restrict__ boxes,   // [TOT_PROP][4]
                                float* __restrict__ nboxes,  // [TOT_PROP][4] (level-offset)
                                float* __restrict__ scores)  // [TOT_PROP]
{
  const int gid = blockIdx.x * blockDim.x + threadIdx.x;
  if (gid >= TOT_PROP) return;
  const int LH[4] = {200, 100, 50, 25};
  const int LPIX[4] = {0, 40000, 50000, 52500};
  const int PO[4] = {0, 120000, 150000, 157500};
  const int STR[4] = {4, 8, 16, 32};
  const int SZ[4] = {32, 64, 128, 256};
  const float ASP[3] = {0.5f, 1.0f, 2.0f};

  int li = 3;
  if (gid < 120000) li = 0; else if (gid < 150000) li = 1; else if (gid < 157500) li = 2;
  const int pi = gid - PO[li];
  const int p = pi / 3, a = pi % 3;
  const int Wd = LH[li];
  const int y = p / Wd, x = p % Wd;

  const v8h* trow = (const v8h*)(t + ((size_t)LPIX[li] + p) * 128);
  float obj = clsb[a];
  float d0 = bbb[a * 4 + 0], d1 = bbb[a * 4 + 1], d2 = bbb[a * 4 + 2], d3 = bbb[a * 4 + 3];
  for (int cc = 0; cc < 16; ++cc) {
    v8h tv8 = trow[cc];
    #pragma unroll
    for (int j = 0; j < 8; ++j) {
      int ch = cc * 8 + j;
      float tv = (float)tv8[j];
      obj += tv * clsw[a * 128 + ch];
      d0 += tv * bbw[(a * 4 + 0) * 128 + ch];
      d1 += tv * bbw[(a * 4 + 1) * 128 + ch];
      d2 += tv * bbw[(a * 4 + 2) * 128 + ch];
      d3 += tv * bbw[(a * 4 + 3) * 128 + ch];
    }
  }
  // anchor
  float hr = sqrtf(ASP[a]);
  float wr = 1.0f / hr;
  float wsz = wr * (float)SZ[li], hsz = hr * (float)SZ[li];
  float ax1 = (float)(x * STR[li]) + roundf(-wsz * 0.5f);
  float ay1 = (float)(y * STR[li]) + roundf(-hsz * 0.5f);
  float ax2 = (float)(x * STR[li]) + roundf(wsz * 0.5f);
  float ay2 = (float)(y * STR[li]) + roundf(hsz * 0.5f);
  // decode (weights 1,1,1,1)
  float aw = ax2 - ax1, ah = ay2 - ay1;
  float cx = ax1 + 0.5f * aw, cy = ay1 + 0.5f * ah;
  float dw = fminf(d2, BBOX_CLIP_F), dh = fminf(d3, BBOX_CLIP_F);
  float pcx = d0 * aw + cx, pcy = d1 * ah + cy;
  float pw = expf(dw) * aw, ph = expf(dh) * ah;
  float bx1 = fminf(fmaxf(pcx - 0.5f * pw, 0.f), IMG_F);
  float by1 = fminf(fmaxf(pcy - 0.5f * ph, 0.f), IMG_F);
  float bx2 = fminf(fmaxf(pcx + 0.5f * pw, 0.f), IMG_F);
  float by2 = fminf(fmaxf(pcy + 0.5f * ph, 0.f), IMG_F);
  boxes[gid * 4 + 0] = bx1; boxes[gid * 4 + 1] = by1;
  boxes[gid * 4 + 2] = bx2; boxes[gid * 4 + 3] = by2;
  const float lo = (float)li * (IMG_F + 1.0f);
  nboxes[gid * 4 + 0] = bx1 + lo; nboxes[gid * 4 + 1] = by1 + lo;
  nboxes[gid * 4 + 2] = bx2 + lo; nboxes[gid * 4 + 3] = by2 + lo;
  float sc = 1.0f / (1.0f + expf(-obj));
  bool ok = (bx2 - bx1 >= 0.001f) && (by2 - by1 >= 0.001f);
  scores[gid] = ok ? sc : NEG_INF;
}

// ---------------------------------------------------------------------------
// Greedy NMS: single workgroup, fused suppress-by-previous + argmax scan.
// ---------------------------------------------------------------------------
__global__ void nms_kernel(const float* __restrict__ boxes, float* __restrict__ scores,
                           int N, int K, float th,
                           int* __restrict__ out_idx, int* __restrict__ out_ok) {
  __shared__ float sv[1024];
  __shared__ int si[1024];
  __shared__ float sel[4];
  const int tid = threadIdx.x;
  if (tid == 0) { sel[0] = sel[1] = sel[2] = sel[3] = 0.f; }
  __syncthreads();
  for (int k = 0; k < K; ++k) {
    const float bx1 = sel[0], by1 = sel[1], bx2 = sel[2], by2 = sel[3];
    const float a1 = (bx2 - bx1) * (by2 - by1);
    float bestv = NEG_INF; int besti = 0;
    for (int j = tid; j < N; j += blockDim.x) {
      float s = scores[j];
      if (s != NEG_INF) {
        if (k > 0) {
          float jx1 = boxes[j * 4], jy1 = boxes[j * 4 + 1];
          float jx2 = boxes[j * 4 + 2], jy2 = boxes[j * 4 + 3];
          float ix1 = fmaxf(bx1, jx1), iy1 = fmaxf(by1, jy1);
          float ix2 = fminf(bx2, jx2), iy2 = fminf(by2, jy2);
          float inter = fmaxf(ix2 - ix1, 0.f) * fmaxf(iy2 - iy1, 0.f);
          float a2 = (jx2 - jx1) * (jy2 - jy1);
          float uni = a1 + a2 - inter;
          float iou = uni > 0.f ? inter / uni : 0.f;
          if (iou > th) { s = NEG_INF; scores[j] = NEG_INF; }
        }
        if (s > bestv || (s == bestv && j < besti)) { bestv = s; besti = j; }
      }
    }
    sv[tid] = bestv; si[tid] = besti;
    __syncthreads();
    for (int off = (int)blockDim.x >> 1; off > 0; off >>= 1) {
      if (tid < off) {
        float ov = sv[tid + off]; int oi = si[tid + off];
        if (ov > sv[tid] || (ov == sv[tid] && oi < si[tid])) { sv[tid] = ov; si[tid] = oi; }
      }
      __syncthreads();
    }
    if (tid == 0) {
      int i = si[0]; float s = sv[0];
      out_idx[k] = i;
      out_ok[k] = (s != NEG_INF) ? 1 : 0;
      sel[0] = boxes[i * 4]; sel[1] = boxes[i * 4 + 1];
      sel[2] = boxes[i * 4 + 2]; sel[3] = boxes[i * 4 + 3];
      scores[i] = NEG_INF;
    }
    __syncthreads();
  }
}

// ---------------------------------------------------------------------------
// Multi-scale ROI-align -> pooled f16 [2000][128*49] (fc6 A layout)
// ---------------------------------------------------------------------------
__global__ void roialign_kernel(const float* __restrict__ f0, const float* __restrict__ f1,
                                const float* __restrict__ f2, const float* __restrict__ f3,
                                const float* __restrict__ boxes, const int* __restrict__ selidx,
                                _Float16* __restrict__ pooled) {
  const int tid = blockIdx.x * blockDim.x + threadIdx.x;
  if (tid >= POST_NMS * 128 * 49) return;
  const int bin = tid % 49;
  const int c = (tid / 49) % 128;
  const int k = tid / (49 * 128);
  const int py = bin / 7, px = bin % 7;
  const int ri = selidx[k];
  const float x1r = boxes[ri * 4 + 0], y1r = boxes[ri * 4 + 1];
  const float x2r = boxes[ri * 4 + 2], y2r = boxes[ri * 4 + 3];
  float area = (x2r - x1r) * (y2r - y1r);
  float tgt = floorf(4.f + log2f(sqrtf(fmaxf(area, 0.f)) / 224.f + 1e-6f));
  int lvl = (int)fminf(fmaxf(tgt, 2.f), 5.f) - 2;
  const int LHa[4] = {200, 100, 50, 25};
  const float SCL[4] = {0.25f, 0.125f, 0.0625f, 0.03125f};
  const float* fp = (lvl == 0) ? f0 : (lvl == 1) ? f1 : (lvl == 2) ? f2 : f3;
  const int H = LHa[lvl], W = LHa[lvl], HW = H * W;
  const float* ch = fp + (size_t)c * HW;
  const float scale = SCL[lvl];
  const float x1 = x1r * scale, y1 = y1r * scale;
  const float rw = fmaxf(x2r * scale - x1, 1.f), rh = fmaxf(y2r * scale - y1, 1.f);
  float sum = 0.f;
  #pragma unroll
  for (int sy = 0; sy < 2; ++sy)
    #pragma unroll
    for (int sx = 0; sx < 2; ++sx) {
      float yv = y1 + ((float)py + (sy + 0.5f) * 0.5f) * (rh / 7.f);
      float xv = x1 + ((float)px + (sx + 0.5f) * 0.5f) * (rw / 7.f);
      bool okv = yv > -1.f && yv < (float)H && xv > -1.f && xv < (float)W;
      float y = fminf(fmaxf(yv, 0.f), (float)H - 1.f);
      float x = fminf(fmaxf(xv, 0.f), (float)W - 1.f);
      int y0 = (int)floorf(y), x0 = (int)floorf(x);
      int y1i = y0 + 1 < H ? y0 + 1 : H - 1;
      int x1i = x0 + 1 < W ? x0 + 1 : W - 1;
      float ly = y - (float)y0, lx = x - (float)x0;
      float hy = 1.f - ly, hx = 1.f - lx;
      float v = hy * hx * ch[y0 * W + x0] + hy * lx * ch[y0 * W + x1i] +
                ly * hx * ch[y1i * W + x0] + ly * lx * ch[y1i * W + x1i];
      sum += okv ? v : 0.f;
    }
  pooled[(size_t)k * 6272 + c * 49 + bin] = (_Float16)(sum * 0.25f);
}

// ---------------------------------------------------------------------------
// cls (2) + bbox (8) linear heads over fc7 output
// ---------------------------------------------------------------------------
__global__ void head_lin_kernel(const _Float16* __restrict__ x,  // [2000][1024]
                                const float* __restrict__ cw, const float* __restrict__ cb,
                                const float* __restrict__ bw, const float* __restrict__ bb,
                                float* __restrict__ ho)          // [2000][10]
{
  const int tid = blockIdx.x * blockDim.x + threadIdx.x;
  if (tid >= POST_NMS * 10) return;
  const int k = tid / 10, o = tid % 10;
  const float* wrow;
  float acc;
  if (o < 2) { wrow = cw + (size_t)o * 1024; acc = cb[o]; }
  else { wrow = bw + (size_t)(o - 2) * 1024; acc = bb[o - 2]; }
  const v8h* xr = (const v8h*)(x + (size_t)k * 1024);
  for (int cc = 0; cc < 128; ++cc) {
    v8h xv = xr[cc];
    #pragma unroll
    for (int j = 0; j < 8; ++j) acc += (float)xv[j] * wrow[cc * 8 + j];
  }
  ho[k * 10 + o] = acc;
}

// ---------------------------------------------------------------------------
// Softmax + box decode (10,10,5,5) + clip + keep mask
// ---------------------------------------------------------------------------
__global__ void head_post_kernel(const float* __restrict__ ho,
                                 const float* __restrict__ rpn_boxes,
                                 const int* __restrict__ selidx, const int* __restrict__ selok,
                                 float* __restrict__ bfg, float* __restrict__ sfg_raw,
                                 float* __restrict__ smask) {
  const int k = blockIdx.x * blockDim.x + threadIdx.x;
  if (k >= POST_NMS) return;
  float l0 = ho[k * 10 + 0], l1 = ho[k * 10 + 1];
  float m = fmaxf(l0, l1);
  float e0 = expf(l0 - m), e1 = expf(l1 - m);
  float s1 = e1 / (e0 + e1);
  const int ri = selidx[k];
  float rx1 = rpn_boxes[ri * 4], ry1 = rpn_boxes[ri * 4 + 1];
  float rx2 = rpn_boxes[ri * 4 + 2], ry2 = rpn_boxes[ri * 4 + 3];
  float w = rx2 - rx1, h = ry2 - ry1;
  float cx = rx1 + 0.5f * w, cy = ry1 + 0.5f * h;
  float d0 = ho[k * 10 + 6] * 0.1f;
  float d1 = ho[k * 10 + 7] * 0.1f;
  float dw = fminf(ho[k * 10 + 8] * 0.2f, BBOX_CLIP_F);
  float dh = fminf(ho[k * 10 + 9] * 0.2f, BBOX_CLIP_F);
  float pcx = d0 * w + cx, pcy = d1 * h + cy;
  float pw = expf(dw) * w, ph = expf(dh) * h;
  float x1 = fminf(fmaxf(pcx - 0.5f * pw, 0.f), IMG_F);
  float y1 = fminf(fmaxf(pcy - 0.5f * ph, 0.f), IMG_F);
  float x2 = fminf(fmaxf(pcx + 0.5f * pw, 0.f), IMG_F);
  float y2 = fminf(fmaxf(pcy + 0.5f * ph, 0.f), IMG_F);
  bfg[k * 4 + 0] = x1; bfg[k * 4 + 1] = y1; bfg[k * 4 + 2] = x2; bfg[k * 4 + 3] = y2;
  sfg_raw[k] = s1;
  bool keep = selok[k] && (s1 > 0.05f) && (x2 - x1 >= 0.01f) && (y2 - y1 >= 0.01f);
  smask[k] = keep ? s1 : NEG_INF;
}

// ---------------------------------------------------------------------------
// Final gather into d_out: ob(2,100,4) f32 | osc(2,100) f32 | ol(2,100) i32
// ---------------------------------------------------------------------------
__global__ void finalize_kernel(const float* __restrict__ bfg, const float* __restrict__ sfg,
                                const int* __restrict__ fidx, const int* __restrict__ fok,
                                float* __restrict__ out) {
  const int tid = blockIdx.x * blockDim.x + threadIdx.x;
  if (tid >= 2 * DETS) return;
  const int b = tid / DETS, i = tid % DETS;
  const int idx = fidx[b * DETS + i];
  const int ok = fok[b * DETS + i];
  const float* bb = bfg + (size_t)b * POST_NMS * 4 + idx * 4;
  float* ob = out + b * DETS * 4 + i * 4;
  ob[0] = ok ? bb[0] : 0.f; ob[1] = ok ? bb[1] : 0.f;
  ob[2] = ok ? bb[2] : 0.f; ob[3] = ok ? bb[3] : 0.f;
  out[2 * DETS * 4 + b * DETS + i] = ok ? sfg[(size_t)b * POST_NMS + idx] : 0.f;
  ((int*)out)[2 * DETS * 4 + 2 * DETS + b * DETS + i] = ok ? 1 : 0;
}

// ---------------------------------------------------------------------------
extern "C" void kernel_launch(void* const* d_in, const int* in_sizes, int n_in,
                              void* d_out, int out_size, void* d_ws, size_t ws_size,
                              hipStream_t stream) {
  const float* feats[4] = {(const float*)d_in[0], (const float*)d_in[1],
                           (const float*)d_in[2], (const float*)d_in[3]};
  const float* rpw  = (const float*)d_in[4];
  const float* rpb  = (const float*)d_in[5];
  const float* rclw = (const float*)d_in[6];
  const float* rclb = (const float*)d_in[7];
  const float* rbw  = (const float*)d_in[8];
  const float* rbb  = (const float*)d_in[9];
  const float* f6w  = (const float*)d_in[10];
  const float* f6b  = (const float*)d_in[11];
  const float* f7w  = (const float*)d_in[12];
  const float* f7b  = (const float*)d_in[13];
  const float* cw   = (const float*)d_in[14];
  const float* cb   = (const float*)d_in[15];
  const float* bw   = (const float*)d_in[16];
  const float* bb   = (const float*)d_in[17];

  const int LH[4] = {200, 100, 50, 25};
  const int LPIX[4] = {0, 40000, 50000, 52500};

  // workspace layout (256B aligned regions)
  char* base = (char*)d_ws;
  size_t off = 0;
  auto alloc = [&](size_t bytes) -> char* {
    char* p = base + off;
    off = (off + bytes + 255) & ~(size_t)255;
    return p;
  };
  _Float16* tin    = (_Float16*)alloc((size_t)2 * TOT_PIX * 128 * 2);  // NHWC features
  _Float16* t      = (_Float16*)alloc((size_t)2 * TOT_PIX * 128 * 2);  // conv output NHWC
  float*    boxes  = (float*)alloc((size_t)2 * TOT_PROP * 4 * 4);
  float*    nboxes = (float*)alloc((size_t)2 * TOT_PROP * 4 * 4);
  float*    scores = (float*)alloc((size_t)2 * TOT_PROP * 4);
  int*      selidx = (int*)alloc((size_t)2 * POST_NMS * 4);
  int*      selok  = (int*)alloc((size_t)2 * POST_NMS * 4);
  _Float16* pooled = (_Float16*)alloc((size_t)2 * POST_NMS * 6272 * 2);
  _Float16* w6h    = (_Float16*)alloc((size_t)1024 * 6272 * 2);
  _Float16* w7h    = (_Float16*)alloc((size_t)1024 * 1024 * 2);
  _Float16* wrh    = (_Float16*)alloc((size_t)9 * 128 * 128 * 2);
  _Float16* zrow   = (_Float16*)alloc((size_t)128 * 2);
  _Float16* x6     = (_Float16*)alloc((size_t)2 * POST_NMS * 1024 * 2);
  _Float16* x7     = (_Float16*)alloc((size_t)2 * POST_NMS * 1024 * 2);
  float*    ho     = (float*)alloc((size_t)2 * POST_NMS * 10 * 4);
  float*    bfg    = (float*)alloc((size_t)2 * POST_NMS * 4 * 4);
  float*    sfgr   = (float*)alloc((size_t)2 * POST_NMS * 4);
  float*    smask  = (float*)alloc((size_t)2 * POST_NMS * 4);
  int*      fidx   = (int*)alloc((size_t)2 * DETS * 4);
  int*      fok    = (int*)alloc((size_t)2 * DETS * 4);
  (void)ws_size; (void)in_sizes; (void)n_in; (void)out_size;

  // weight conversions + zero row (once per call, deterministic)
  cvt_f32_f16<<<(6272 * 1024 + 255) / 256, 256, 0, stream>>>(f6w, w6h, 6272 * 1024);
  cvt_f32_f16<<<(1024 * 1024 + 255) / 256, 256, 0, stream>>>(f7w, w7h, 1024 * 1024);
  cvt_rpnw<<<(9 * 128 * 128 + 255) / 256, 256, 0, stream>>>(rpw, wrh);
  zero_row_f16<<<1, 128, 0, stream>>>(zrow);

  for (int b = 0; b < 2; ++b) {
    // NCHW f32 -> NHWC f16 per level
    for (int li = 0; li < 4; ++li) {
      int HW = LH[li] * LH[li];
      cvt_feat_nhwc<<<(128 * HW + 255) / 256, 256, 0, stream>>>(
          feats[li] + (size_t)b * 128 * HW,
          tin + ((size_t)b * TOT_PIX + LPIX[li]) * 128, HW);
    }
    // RPN conv per level (WMMA)
    for (int li = 0; li < 4; ++li) {
      int H = LH[li], W = LH[li], HW = H * W;
      int ptiles = (HW + 15) / 16;
      int threads = ptiles * 8 * 32;
      rpn_conv3x3_wmma<<<(threads + 255) / 256, 256, 0, stream>>>(
          tin + ((size_t)b * TOT_PIX + LPIX[li]) * 128, wrh, rpb, zrow,
          t + ((size_t)b * TOT_PIX + LPIX[li]) * 128, H, W);
    }
    // proposals
    proposal_kernel<<<(TOT_PROP + 255) / 256, 256, 0, stream>>>(
        t + (size_t)b * TOT_PIX * 128, rclw, rclb, rbw, rbb,
        boxes + (size_t)b * TOT_PROP * 4, nboxes + (size_t)b * TOT_PROP * 4,
        scores + (size_t)b * TOT_PROP);
    // RPN NMS -> 2000 rois
    nms_kernel<<<1, 1024, 0, stream>>>(
        nboxes + (size_t)b * TOT_PROP * 4, scores + (size_t)b * TOT_PROP,
        TOT_PROP, POST_NMS, 0.7f, selidx + b * POST_NMS, selok + b * POST_NMS);
    // ROI align
    roialign_kernel<<<(POST_NMS * 128 * 49 + 255) / 256, 256, 0, stream>>>(
        feats[0] + (size_t)b * 128 * 40000, feats[1] + (size_t)b * 128 * 10000,
        feats[2] + (size_t)b * 128 * 2500, feats[3] + (size_t)b * 128 * 625,
        boxes + (size_t)b * TOT_PROP * 4, selidx + b * POST_NMS,
        pooled + (size_t)b * POST_NMS * 6272);
  }

  // fc6 / fc7 WMMA GEMMs
  for (int b = 0; b < 2; ++b) {
    int waves = (POST_NMS / 16) * (1024 / 16);
    gemm_wmma_f16<<<(waves * 32 + 255) / 256, 256, 0, stream>>>(
        pooled + (size_t)b * POST_NMS * 6272, w6h, f6b,
        x6 + (size_t)b * POST_NMS * 1024, POST_NMS, 1024, 6272, 1);
  }
  for (int b = 0; b < 2; ++b) {
    int waves = (POST_NMS / 16) * (1024 / 16);
    gemm_wmma_f16<<<(waves * 32 + 255) / 256, 256, 0, stream>>>(
        x6 + (size_t)b * POST_NMS * 1024, w7h, f7b,
        x7 + (size_t)b * POST_NMS * 1024, POST_NMS, 1024, 1024, 1);
  }

  for (int b = 0; b < 2; ++b) {
    head_lin_kernel<<<(POST_NMS * 10 + 255) / 256, 256, 0, stream>>>(
        x7 + (size_t)b * POST_NMS * 1024, cw, cb, bw, bb, ho + (size_t)b * POST_NMS * 10);
    head_post_kernel<<<(POST_NMS + 255) / 256, 256, 0, stream>>>(
        ho + (size_t)b * POST_NMS * 10, boxes + (size_t)b * TOT_PROP * 4,
        selidx + b * POST_NMS, selok + b * POST_NMS,
        bfg + (size_t)b * POST_NMS * 4, sfgr + (size_t)b * POST_NMS,
        smask + (size_t)b * POST_NMS);
    nms_kernel<<<1, 1024, 0, stream>>>(
        bfg + (size_t)b * POST_NMS * 4, smask + (size_t)b * POST_NMS,
        POST_NMS, DETS, 0.5f, fidx + b * DETS, fok + b * DETS);
  }

  finalize_kernel<<<1, 256, 0, stream>>>(bfg, sfgr, fidx, fok, (float*)d_out);
}
